// Model_7301444403487
// MI455X (gfx1250) — matrix operand order repaired and checked by reading, not compiled, not run on hardware
//
#include <hip/hip_runtime.h>
#include <hip/hip_bf16.h>

typedef __attribute__((ext_vector_type(16))) _Float16 v16h;
typedef __attribute__((ext_vector_type(8)))  _Float16 v8h;
typedef __attribute__((ext_vector_type(8)))  float    v8f;

#define DOUT 256          // output channels of both layers
#define LDA  40           // padded LDS stride in f16 (80B: 16B aligned, conflict-free operand reads)

// ---------------------------------------------------------------------------
// Degree / norm kernels
// ---------------------------------------------------------------------------
__global__ __launch_bounds__(256) void deg_init_kernel(float* deg, int n) {
    int i = blockIdx.x * 256 + threadIdx.x;
    if (i < n) deg[i] = 1.0f;   // implicit self-loop
}

__global__ __launch_bounds__(256) void deg_count_kernel(const int* __restrict__ dst,
                                                        float* deg, int E) {
    int e = blockIdx.x * 256 + threadIdx.x;
    if (e < E) unsafeAtomicAdd(&deg[dst[e]], 1.0f);
}

__global__ __launch_bounds__(256) void deg_rsqrt_kernel(float* deg, int n) {
    int i = blockIdx.x * 256 + threadIdx.x;
    if (i < n) deg[i] = rsqrtf(deg[i]);   // in place: deg[] now holds d^{-1/2}
}

// ---------------------------------------------------------------------------
// Wt[n][k] = (f16) W[k][n]   — one-time per layer, W is <=512KB (L2 resident)
// ---------------------------------------------------------------------------
__global__ __launch_bounds__(256) void transpose_w_kernel(
    const float* __restrict__ W, _Float16* __restrict__ Wt, int K)
{
    int idx = blockIdx.x * 256 + threadIdx.x;   // consecutive idx: same n, consecutive k
    if (idx >= K * DOUT) return;
    int n = idx / K;
    int k = idx - n * K;
    Wt[idx] = (_Float16)W[(size_t)k * DOUT + n];
}

// ---------------------------------------------------------------------------
// H = A(MxK f32, optional compile-time relu) @ W(Kx256)  via v_wmma_f32_16x16x32_f16
// Block: 256 threads = 8 waves, block tile 128(M) x 64(N).
// Wave w owns rows [w*16, w*16+16) x all 64 cols: 4 accumulators sharing one A operand.
// Double-buffered LDS: 1 barrier per K-step; next tile's global loads overlap the wmmas.
// ---------------------------------------------------------------------------
template<int RELU>
__global__ __launch_bounds__(256) void gemm_f16_wmma_kernel(
    const float* __restrict__ A, const _Float16* __restrict__ Wt,
    float* __restrict__ Hout, int M, int K)
{
    __shared__ __align__(16) _Float16 Alds[2][128 * LDA];
    __shared__ __align__(16) _Float16 Blds[2][64 * LDA];

    const int t    = threadIdx.x;
    const int lane = t & 31;
    const int wave = t >> 5;                 // 0..7 -> M sub-row
    const int mBase = blockIdx.x * 128;
    const int nBase = blockIdx.y * 64;

    v8f acc0 = {}, acc1 = {}, acc2 = {}, acc3 = {};

    const int srow = t >> 2;                 // 0..63 (staging row base)
    const int skc  = (t & 3) * 8;            // 0,8,16,24
    const int grow0 = mBase + srow;
    const int grow1 = mBase + srow + 64;
    const bool g0ok = grow0 < M;
    const bool g1ok = grow1 < M;

    const int r     = lane & 15;             // row (A) / col (B) within a 16-tile
    const int khalf = lane >> 4;             // 0 or 1

    float4 ra[4];                            // raw A fetch: 2 rows x 8 f32
    v8h    rb;                               // B fetch (already f16)

    // issue global fetches for K-tile at k0
    auto fetch = [&](int k0) {
        if (g0ok) {
            const float4* sp = (const float4*)&A[(size_t)grow0 * K + k0 + skc];
            ra[0] = sp[0]; ra[1] = sp[1];
        }
        if (g1ok) {
            const float4* sp = (const float4*)&A[(size_t)grow1 * K + k0 + skc];
            ra[2] = sp[0]; ra[3] = sp[1];
        }
        rb = *(const v8h*)&Wt[(size_t)(nBase + srow) * K + k0 + skc];
    };

    // convert + store fetched tile into LDS buffer `buf`
    auto cvtstore = [&](int buf) {
        #pragma unroll
        for (int h = 0; h < 2; ++h) {
            v8h a16 = {};
            if (h == 0 ? g0ok : g1ok) {
                float tmp[8] = {ra[2*h].x, ra[2*h].y, ra[2*h].z, ra[2*h].w,
                                ra[2*h+1].x, ra[2*h+1].y, ra[2*h+1].z, ra[2*h+1].w};
                #pragma unroll
                for (int j = 0; j < 8; ++j) {
                    float v = tmp[j];
                    if (RELU) v = fmaxf(v, 0.0f);
                    a16[j] = (_Float16)v;
                }
            }
            *(v8h*)&Alds[buf][(srow + 64 * h) * LDA + skc] = a16;
        }
        *(v8h*)&Blds[buf][srow * LDA + skc] = rb;
    };

    // prologue: stage tile 0
    fetch(0);
    cvtstore(0);
    __syncthreads();

    for (int k0 = 0; k0 < K; k0 += 32) {
        const int  buf  = (k0 >> 5) & 1;
        const bool more = (k0 + 32) < K;

        if (more) fetch(k0 + 32);            // global loads in flight during compute

        // ---- A operand (ISA 16-bit A 16x32 layout) ----
        // lane l: row = l&15; elems 0..7 = K[kb..kb+7], 8..15 = K[kb+16..kb+23], kb=(l>>4)*8
        const _Float16* ap = &Alds[buf][(wave * 16 + r) * LDA + khalf * 8];
        v8h alo = *(const v8h*)(ap);
        v8h ahi = *(const v8h*)(ap + 16);
        v16h av;
        #pragma unroll
        for (int j = 0; j < 8; ++j) { av[j] = alo[j]; av[j + 8] = ahi[j]; }

        // ---- preload all four B operands, then wmma back-to-back ----
        // B operand: lane l: col = l&15; elems 0..7 = K[kb..kb+7], 8..15 = K[kb+16..kb+23], kb=(l>>4)*8
        v16h bv[4];
        #pragma unroll
        for (int s = 0; s < 4; ++s) {
            const _Float16* bp = &Blds[buf][(s * 16 + r) * LDA + khalf * 8];
            v8h blo = *(const v8h*)(bp);
            v8h bhi = *(const v8h*)(bp + 16);
            #pragma unroll
            for (int j = 0; j < 8; ++j) { bv[s][j] = blo[j]; bv[s][j + 8] = bhi[j]; }
        }
        acc0 = __builtin_amdgcn_wmma_f32_16x16x32_f16(false, av, false, bv[0],
                                                      (short)0, acc0, false, false);
        acc1 = __builtin_amdgcn_wmma_f32_16x16x32_f16(false, av, false, bv[1],
                                                      (short)0, acc1, false, false);
        acc2 = __builtin_amdgcn_wmma_f32_16x16x32_f16(false, av, false, bv[2],
                                                      (short)0, acc2, false, false);
        acc3 = __builtin_amdgcn_wmma_f32_16x16x32_f16(false, av, false, bv[3],
                                                      (short)0, acc3, false, false);

        if (more) {
            cvtstore(buf ^ 1);               // fill the other buffer
            __syncthreads();                 // single barrier per K-step
        }
    }

    // ---- store C (lane l: N = l&15, VGPR v: M = v + 8*(l>>4)) ----
    const int hm = lane >> 4;
    #pragma unroll
    for (int v = 0; v < 8; ++v) {
        int row = mBase + wave * 16 + hm * 8 + v;
        if (row < M) {
            float* op = &Hout[(size_t)row * DOUT + nBase + r];
            op[0]  = acc0[v];
            op[16] = acc1[v];
            op[32] = acc2[v];
            op[48] = acc3[v];
        }
    }
}

// ---------------------------------------------------------------------------
// out[i][c] = H[i][c] * dinv[i]^2 + b[c]      (self-loop term + bias, init pass)
// ---------------------------------------------------------------------------
__global__ __launch_bounds__(256) void self_bias_kernel(
    const float* __restrict__ H, const float* __restrict__ dinv,
    const float* __restrict__ b, float* __restrict__ out, int n)
{
    int tid  = blockIdx.x * 256 + threadIdx.x;
    int node = tid >> 6;
    int c    = (tid & 63) * 4;
    if (node >= n) return;
    float di = dinv[node];
    float s  = di * di;               // 1/deg
    const float4 h = *(const float4*)&H[(size_t)node * DOUT + c];
    float4 bb      = *(const float4*)&b[c];
    float4 o;
    o.x = h.x * s + bb.x;
    o.y = h.y * s + bb.y;
    o.z = h.z * s + bb.z;
    o.w = h.w * s + bb.w;
    *(float4*)&out[(size_t)node * DOUT + c] = o;
}

// ---------------------------------------------------------------------------
// out[dst[e]][c] += H[src[e]][c] * dinv[src]*dinv[dst]
// one thread = (edge, 4 channels); H rows are L2-resident (51 MB < 192 MB L2)
// ---------------------------------------------------------------------------
__global__ __launch_bounds__(256) void edge_scatter_kernel(
    const float* __restrict__ H, const int* __restrict__ src,
    const int* __restrict__ dst, const float* __restrict__ dinv,
    float* __restrict__ out, int E)
{
    int tid = blockIdx.x * 256 + threadIdx.x;
    int e   = tid >> 6;
    int c   = (tid & 63) * 4;
    if (e >= E) return;
    int s = src[e];
    int d = dst[e];
    float norm = dinv[s] * dinv[d];
    const float4 h = *(const float4*)&H[(size_t)s * DOUT + c];
    float* o = &out[(size_t)d * DOUT + c];
    unsafeAtomicAdd(o + 0, h.x * norm);
    unsafeAtomicAdd(o + 1, h.y * norm);
    unsafeAtomicAdd(o + 2, h.z * norm);
    unsafeAtomicAdd(o + 3, h.w * norm);
}

// ---------------------------------------------------------------------------
// Launch: two views, each = 2 GCN layers (GEMM -> self+bias -> edge scatter)
// ---------------------------------------------------------------------------
extern "C" void kernel_launch(void* const* d_in, const int* in_sizes, int n_in,
                              void* d_out, int out_size, void* d_ws, size_t ws_size,
                              hipStream_t stream)
{
    const int IN_DIM = 512, HID = 256;
    const int N = in_sizes[0] / IN_DIM;     // 50000
    const int E = in_sizes[2] / 2;          // 800000

    const float* xs[2]  = { (const float*)d_in[0], (const float*)d_in[1] };
    const int*   eis[2] = { (const int*)d_in[2],   (const int*)d_in[3]   };
    const float* W1 = (const float*)d_in[4];
    const float* b1 = (const float*)d_in[5];
    const float* W2 = (const float*)d_in[6];
    const float* b2 = (const float*)d_in[7];
    float* out = (float*)d_out;

    // workspace: deg/dinv | H (Nx256 f32) | AGG (Nx256 f32) | Wt (256xK f16)
    float* deg = (float*)d_ws;
    size_t degPad = ((size_t)N + 255) & ~(size_t)255;
    float*    Hbuf   = deg + degPad;
    float*    AGGbuf = Hbuf + (size_t)N * DOUT;
    _Float16* Wtbuf  = (_Float16*)(AGGbuf + (size_t)N * DOUT);  // 256*512 f16 max

    const int nodeBlocks = (N + 255) / 256;
    const int edgeBlocks = (E + 255) / 256;
    const int nc4Blocks  = (int)(((size_t)N * 64 + 255) / 256);
    const int ec4Blocks  = (int)(((size_t)E * 64 + 255) / 256);
    const dim3 gemmGrid((N + 127) / 128, DOUT / 64);
    const int wt1Blocks = (IN_DIM * DOUT + 255) / 256;
    const int wt2Blocks = (HID   * DOUT + 255) / 256;

    for (int view = 0; view < 2; ++view) {
        const float* x   = xs[view];
        const int*   src = eis[view];
        const int*   dst = eis[view] + E;
        float* outv = out + (size_t)view * N * DOUT;

        // normalization factors
        deg_init_kernel <<<nodeBlocks, 256, 0, stream>>>(deg, N);
        deg_count_kernel<<<edgeBlocks, 256, 0, stream>>>(dst, deg, E);
        deg_rsqrt_kernel<<<nodeBlocks, 256, 0, stream>>>(deg, N);

        // layer 1: H = x @ W1 ; AGG = norm-agg(H) + b1
        transpose_w_kernel     <<<wt1Blocks, 256, 0, stream>>>(W1, Wtbuf, IN_DIM);
        gemm_f16_wmma_kernel<0><<<gemmGrid, 256, 0, stream>>>(x, Wtbuf, Hbuf, N, IN_DIM);
        self_bias_kernel       <<<nc4Blocks, 256, 0, stream>>>(Hbuf, deg, b1, AGGbuf, N);
        edge_scatter_kernel    <<<ec4Blocks, 256, 0, stream>>>(Hbuf, src, dst, deg, AGGbuf, E);

        // layer 2: H = relu(AGG) @ W2 ; out = norm-agg(H) + b2
        transpose_w_kernel     <<<wt2Blocks, 256, 0, stream>>>(W2, Wtbuf, HID);
        gemm_f16_wmma_kernel<1><<<gemmGrid, 256, 0, stream>>>(AGGbuf, Wtbuf, Hbuf, N, HID);
        self_bias_kernel       <<<nc4Blocks, 256, 0, stream>>>(Hbuf, deg, b2, outv, N);
        edge_scatter_kernel    <<<ec4Blocks, 256, 0, stream>>>(Hbuf, src, dst, deg, outv, E);
    }
}
